// SlidingWindowPyramidAttention_37452114821896
// MI455X (gfx1250) — matrix operand order repaired and checked by reading, not compiled, not run on hardware
//
#include <hip/hip_runtime.h>
#include <hip/hip_bf16.h>

// ---------------- problem constants ----------------
#define BB 4
#define CC 128
#define HH 128
#define WW 128
#define HWH (HH * WW)           // 16384
#define NH 8
#define NP 4
#define NL 3
#define HD 16                   // C / NH
#define C2 64
#define GG 8

typedef __attribute__((ext_vector_type(16))) __bf16 v16bf;
typedef __attribute__((ext_vector_type(8)))  __bf16 bf16x8;
typedef __attribute__((ext_vector_type(8)))  float  v8f;

union FragBF { v16bf v; bf16x8 h[2]; };

// =====================================================================
// Generic 1x1-conv GEMM:  Y[z, o, p] = sum_c W[o,c]*(X[z,c,p]+add[c]) + bias[o]
// CIN in {64,128} (template), Cout runtime multiple of 16.
// LAYOUT 0: Y stored channel-major  Yz[o*HW + p]
// LAYOUT 1: Y stored head-interleaved Yz[(mt*HW + p)*16 + hd]   (Cout==128)
// Block: 256 threads (8 waves). Tile: 64 pixels x Cout outputs.
// Uses v_wmma_f32_16x16x32_bf16 with fp32 accumulation.
// =====================================================================
template <int CIN, int LAYOUT>
__global__ __launch_bounds__(256)
void conv1x1_wmma_kernel(const float* __restrict__ X, long xStrideZ,
                         const float* __restrict__ Wm,
                         const float* __restrict__ bias,
                         const float* __restrict__ addvec, int addMod,
                         float* __restrict__ Y, long yStrideZ, int Cout)
{
    constexpr int CINP = CIN + 8;           // padded row (bytes multiple of 16)
    extern __shared__ char smem_raw[];
    __bf16* Ws = (__bf16*)smem_raw;                 // [Cout][CINP]
    __bf16* Xs = Ws + (size_t)Cout * CINP;          // [64][CINP]

    const int tid = threadIdx.x;
    const int z   = blockIdx.y;
    const int p0  = blockIdx.x * 64;

    const float* Xz = X + (long)z * xStrideZ;
    float*       Yz = Y + (long)z * yStrideZ;
    const float* av = addvec ? (addvec + (z % addMod) * CIN) : nullptr;

    // ---- stage weights (row-major [Cout][CIN]) -> bf16 LDS ----
    for (int idx = tid; idx < Cout * CIN; idx += 256) {
        int o = idx / CIN, k = idx - o * CIN;
        Ws[o * CINP + k] = (__bf16)Wm[idx];
    }
    // ---- stage X tile transposed: [channel-major global] -> [pixel][channel] bf16 ----
    for (int idx = tid; idx < CIN * 64; idx += 256) {
        int c = idx >> 6, pl = idx & 63;
        float v = Xz[(long)c * HWH + p0 + pl];
        if (av) v += av[c];
        Xs[pl * CINP + c] = (__bf16)v;
    }
    // prefetch next pixel tile (global_prefetch_b8)
    if (blockIdx.x + 1 < gridDim.x) {
        for (int c = tid; c < CIN; c += 256)
            __builtin_prefetch(Xz + (long)c * HWH + p0 + 64, 0, 1);
    }
    __syncthreads();

    const int wave  = tid >> 5;
    const int lane  = tid & 31;
    const int ln    = lane & 15;
    const int lhalf = (lane < 16) ? 0 : 8;      // A-frag K sub-block select
    const int bkoff = (lane < 16) ? 0 : 16;     // (unused) B-frag uses lhalf like A
    const int nMT   = Cout >> 4;

    for (int mt = wave; mt < nMT; mt += 8) {
        v8f acc0 = {}, acc1 = {}, acc2 = {}, acc3 = {};
#pragma unroll
        for (int kc = 0; kc < CIN / 32; ++kc) {
            // A fragment: 16x32 of W (M = out channel, K = in channel)
            FragBF a;
            const __bf16* wrow = Ws + (mt * 16 + ln) * CINP + kc * 32;
            a.h[0] = *(const bf16x8*)(wrow + lhalf);
            a.h[1] = *(const bf16x8*)(wrow + 16 + lhalf);
            // B fragments: 32x16 of X (K = in channel, N = pixel)
#pragma unroll
            for (int nt = 0; nt < 4; ++nt) {
                FragBF bfr;
                const __bf16* xrow = Xs + (nt * 16 + ln) * CINP + kc * 32 + lhalf;
                bfr.h[0] = *(const bf16x8*)(xrow);
                bfr.h[1] = *(const bf16x8*)(xrow + 16);
                v8f* accp = (nt == 0) ? &acc0 : (nt == 1) ? &acc1 : (nt == 2) ? &acc2 : &acc3;
                *accp = __builtin_amdgcn_wmma_f32_16x16x32_bf16(
                            false, a.v, false, bfr.v, (short)0, *accp, false, false);
            }
        }
        // ---- epilogue: bias + store ----
#pragma unroll
        for (int nt = 0; nt < 4; ++nt) {
            v8f acc = (nt == 0) ? acc0 : (nt == 1) ? acc1 : (nt == 2) ? acc2 : acc3;
            int p = p0 + nt * 16 + ln;
#pragma unroll
            for (int i = 0; i < 8; ++i) {
                int o = mt * 16 + i + lhalf;
                float v = acc[i] + bias[o];
                if (LAYOUT == 0) Yz[(long)o * HWH + p] = v;
                else             Yz[((long)mt * HWH + p) * 16 + i + lhalf] = v;
            }
        }
    }
}

// =====================================================================
// GroupNorm statistics: one block per (batch, group-of-8-channels) over
// the 128-channel packed hidden (so: ch 0..63 -> groups 0..7,
// aw: ch 64..127 -> groups 8..15). stats[bg] = {mean, rstd}.
// =====================================================================
__global__ __launch_bounds__(256)
void gn_stats_kernel(const float* __restrict__ hid, float* __restrict__ stats)
{
    int bg = blockIdx.x;                 // 0..63
    int b = bg >> 4, g = bg & 15;
    const float* base = hid + ((long)b * 128 + g * 8) * HWH;
    float s = 0.f, ss = 0.f;
    for (int idx = threadIdx.x; idx < 8 * HWH; idx += 256) {
        float v = base[(long)(idx >> 14) * HWH + (idx & (HWH - 1))];
        s += v; ss += v * v;
    }
    __shared__ float rs[256], rss[256];
    rs[threadIdx.x] = s; rss[threadIdx.x] = ss;
    __syncthreads();
    for (int off = 128; off > 0; off >>= 1) {
        if (threadIdx.x < off) {
            rs[threadIdx.x]  += rs[threadIdx.x + off];
            rss[threadIdx.x] += rss[threadIdx.x + off];
        }
        __syncthreads();
    }
    if (threadIdx.x == 0) {
        const float invN = 1.0f / (8.0f * HWH);
        float mean = rs[0] * invN;
        float var  = rss[0] * invN - mean * mean;
        stats[bg * 2 + 0] = mean;
        stats[bg * 2 + 1] = rsqrtf(var + 1e-5f);
    }
}

// GroupNorm apply + affine + ReLU, in place on packed hidden [B][128][HW].
__global__ __launch_bounds__(256)
void gn_apply_kernel(float* __restrict__ hid, const float* __restrict__ stats,
                     const float* __restrict__ so_g, const float* __restrict__ so_be,
                     const float* __restrict__ aw_g, const float* __restrict__ aw_be)
{
    long idx = (long)blockIdx.x * 256 + threadIdx.x;
    const long total = (long)BB * 128 * HWH;
    if (idx >= total) return;
    int c  = (int)((idx >> 14) & 127);
    int b  = (int)(idx >> 21);               // idx / (128*16384)
    int bg = b * 16 + (c >> 3);
    float mu = stats[bg * 2 + 0], rstd = stats[bg * 2 + 1];
    float gam = (c < 64) ? so_g[c] : aw_g[c - 64];
    float bet = (c < 64) ? so_be[c] : aw_be[c - 64];
    float v = (hid[idx] - mu) * rstd * gam + bet;
    hid[idx] = v > 0.f ? v : 0.f;
}

// Softmax over the 12 (level,point) logits per (b, pixel, head), in place.
__global__ __launch_bounds__(256)
void softmax_kernel(float* __restrict__ attn)
{
    int idx = blockIdx.x * 256 + threadIdx.x;       // b*NH*HW threads
    if (idx >= BB * NH * HWH) return;
    int p  = idx & (HWH - 1);
    int t  = idx >> 14;
    int nh = t & 7, b = t >> 3;
    float* base = attn + ((long)b * 96 + nh * 12) * HWH + p;
    float v[12], m = -1e30f;
#pragma unroll
    for (int j = 0; j < 12; ++j) { v[j] = base[(long)j * HWH]; m = fmaxf(m, v[j]); }
    float s = 0.f;
#pragma unroll
    for (int j = 0; j < 12; ++j) { v[j] = __expf(v[j] - m); s += v[j]; }
    float inv = 1.0f / s;
#pragma unroll
    for (int j = 0; j < 12; ++j) base[(long)j * HWH] = v[j] * inv;
}

// =====================================================================
// Deformable bilinear sampling + attention-weighted accumulation.
// Block = (b, head, 16-pixel tile). Phase 1: 192 threads compute the 12
// sample coords/weights per pixel into LDS. Phase 2: 256 threads
// (pixel x head-dim) gather 4 taps per sample from v_proj laid out as
// [b][lvl][head][pixel][16 hd] (contiguous 64B per tap for 16 lanes).
// =====================================================================
__global__ __launch_bounds__(256)
void sample_kernel(const float* __restrict__ off_raw,
                   const float* __restrict__ attn,
                   const float* __restrict__ vproj,
                   float* __restrict__ out_mid)
{
    const int b = blockIdx.z, nh = blockIdx.y, p0 = blockIdx.x * 16;
    __shared__ int   sx0[16][12], sy0[16][12], sx1[16][12], sy1[16][12];
    __shared__ float swx[16][12], swy[16][12], sa[16][12];
    const int tid = threadIdx.x;

    if (tid < 192) {
        int jp = tid / 12, j = tid - jp * 12;
        int p = p0 + jp;
        int lvl = j >> 2, pt = j & 3;
        int ch = ((nh * NL + lvl) * NP + pt) * 2;
        const float* ob = off_raw + (long)b * 192 * HWH + p;
        float ox = ob[(long)ch * HWH];
        float oy = ob[(long)(ch + 1) * HWH];
        int py_ = p >> 7, px_ = p & 127;
        float lx = px_ * (1.0f / (WW - 1)) + ox * (1.0f / WW);
        float ly = py_ * (1.0f / (HH - 1)) + oy * (1.0f / HH);
        lx = fminf(fmaxf(lx, 0.f), 1.f);
        ly = fminf(fmaxf(ly, 0.f), 1.f);
        float pxs = fminf(fmaxf(lx * WW - 0.5f, 0.f), (float)(WW - 1));
        float pys = fminf(fmaxf(ly * HH - 0.5f, 0.f), (float)(HH - 1));
        float x0f = floorf(pxs), y0f = floorf(pys);
        int x0i = (int)x0f, y0i = (int)y0f;
        swx[jp][j] = pxs - x0f; swy[jp][j] = pys - y0f;
        sx0[jp][j] = x0i;                sy0[jp][j] = y0i;
        sx1[jp][j] = min(x0i + 1, WW-1); sy1[jp][j] = min(y0i + 1, HH-1);
        sa[jp][j]  = attn[((long)b * 96 + nh * 12 + j) * HWH + p];
    }
    __syncthreads();

    int jp = tid >> 4, hd = tid & 15;
    int p = p0 + jp;
    float accv = 0.f;
#pragma unroll
    for (int j = 0; j < 12; ++j) {
        int lvl = j >> 2;
        const float* vb = vproj + ((((long)(b * NL + lvl) * NH + nh) * HWH) << 4) + hd;
        int x0 = sx0[jp][j], x1 = sx1[jp][j], y0 = sy0[jp][j], y1 = sy1[jp][j];
        float wx = swx[jp][j], wy = swy[jp][j], a = sa[jp][j];
        float v00 = vb[((long)(y0 * WW + x0)) << 4];
        float v01 = vb[((long)(y0 * WW + x1)) << 4];
        float v10 = vb[((long)(y1 * WW + x0)) << 4];
        float v11 = vb[((long)(y1 * WW + x1)) << 4];
        float top = v00 + (v01 - v00) * wx;
        float bot = v10 + (v11 - v10) * wx;
        accv += a * (top + (bot - top) * wy);
    }
    out_mid[((long)b * 128 + nh * 16 + hd) * HWH + p] = accv;
}

// =====================================================================
extern "C" void kernel_launch(void* const* d_in, const int* in_sizes, int n_in,
                              void* d_out, int out_size, void* d_ws, size_t ws_size,
                              hipStream_t stream)
{
    const float* query  = (const float*)d_in[0];
    // d_in[1] (keys) unused by the reference computation
    const float* values = (const float*)d_in[2];
    const float* so_w1  = (const float*)d_in[3];
    const float* so_b1  = (const float*)d_in[4];
    const float* so_g   = (const float*)d_in[5];
    const float* so_be  = (const float*)d_in[6];
    const float* so_w2  = (const float*)d_in[7];
    const float* so_b2  = (const float*)d_in[8];
    const float* aw_w1  = (const float*)d_in[9];
    const float* aw_b1  = (const float*)d_in[10];
    const float* aw_g   = (const float*)d_in[11];
    const float* aw_be  = (const float*)d_in[12];
    const float* aw_w2  = (const float*)d_in[13];
    const float* aw_b2  = (const float*)d_in[14];
    const float* vp_w   = (const float*)d_in[15];
    const float* vp_b   = (const float*)d_in[16];
    const float* op_w   = (const float*)d_in[17];
    const float* op_b   = (const float*)d_in[18];
    const float* lvlemb = (const float*)d_in[19];
    float* out = (float*)d_out;

    // ---- workspace partition (floats) ----
    float* hid      = (float*)d_ws;                    //  B*128*HW  =  8,388,608
    float* stats    = hid      + (long)BB * 128 * HWH; //  128
    float* off_raw  = stats    + 128;                  //  B*192*HW  = 12,582,912
    float* attn_b   = off_raw  + (long)BB * 192 * HWH; //  B*96*HW   =  6,291,456
    float* vproj    = attn_b   + (long)BB * 96 * HWH;  //  B*NL*C*HW = 25,165,824
    float* out_mid  = vproj    + (long)BB * NL * CC * HWH; // B*C*HW =  8,388,608

    const dim3 blk(256);
    const int ptiles = HWH / 64;   // 256

    // MLP layer-1 for both heads -> packed hidden [B][128][HW]
    size_t sm_128_64 = ((size_t)64 * (128 + 8) + (size_t)64 * (128 + 8)) * 2;
    conv1x1_wmma_kernel<128, 0><<<dim3(ptiles, BB), blk, sm_128_64, stream>>>(
        query, (long)CC * HWH, so_w1, so_b1, nullptr, 1,
        hid, (long)128 * HWH, 64);
    conv1x1_wmma_kernel<128, 0><<<dim3(ptiles, BB), blk, sm_128_64, stream>>>(
        query, (long)CC * HWH, aw_w1, aw_b1, nullptr, 1,
        hid + (long)64 * HWH, (long)128 * HWH, 64);

    // GroupNorm + ReLU
    gn_stats_kernel<<<dim3(BB * 16), blk, 0, stream>>>(hid, stats);
    {
        long total = (long)BB * 128 * HWH;
        gn_apply_kernel<<<dim3((unsigned)((total + 255) / 256)), blk, 0, stream>>>(
            hid, stats, so_g, so_be, aw_g, aw_be);
    }

    // MLP layer-2: offsets (192ch) and attention logits (96ch)
    size_t sm_64_192 = ((size_t)192 * (64 + 8) + (size_t)64 * (64 + 8)) * 2;
    size_t sm_64_96  = ((size_t)96  * (64 + 8) + (size_t)64 * (64 + 8)) * 2;
    conv1x1_wmma_kernel<64, 0><<<dim3(ptiles, BB), blk, sm_64_192, stream>>>(
        hid, (long)128 * HWH, so_w2, so_b2, nullptr, 1,
        off_raw, (long)192 * HWH, 192);
    conv1x1_wmma_kernel<64, 0><<<dim3(ptiles, BB), blk, sm_64_96, stream>>>(
        hid + (long)64 * HWH, (long)128 * HWH, aw_w2, aw_b2, nullptr, 1,
        attn_b, (long)96 * HWH, 96);

    // Softmax over 12 per (b, pixel, head)
    softmax_kernel<<<dim3((BB * NH * HWH + 255) / 256), blk, 0, stream>>>(attn_b);

    // Value projection (+level embed) for all (b, lvl), head-interleaved layout
    size_t sm_128_128 = ((size_t)128 * (128 + 8) + (size_t)64 * (128 + 8)) * 2;
    conv1x1_wmma_kernel<128, 1><<<dim3(ptiles, BB * NL), blk, sm_128_128, stream>>>(
        values, (long)CC * HWH, vp_w, vp_b, lvlemb, NL,
        vproj, (long)CC * HWH, 128);

    // Deformable bilinear sampling + attention weighting
    sample_kernel<<<dim3(HWH / 16, NH, BB), blk, 0, stream>>>(
        off_raw, attn_b, vproj, out_mid);

    // Output projection -> d_out [B][C][H][W]
    conv1x1_wmma_kernel<128, 0><<<dim3(ptiles, BB), blk, sm_128_128, stream>>>(
        out_mid, (long)CC * HWH, op_w, op_b, nullptr, 1,
        out, (long)CC * HWH, 128);
}
